// ContrastLoss_7842610283314
// MI455X (gfx1250) — compile-verified, matrix operand order checked by reading, not run on hardware
//
#include <hip/hip_runtime.h>
#include <hip/hip_bf16.h>

typedef __attribute__((ext_vector_type(16))) _Float16 v16h;
typedef __attribute__((ext_vector_type(8)))  float    v8f;

#define BINS      256
#define TILE_PIX  32
#define ROW_H     40        // halves per tile row: 32 pixels (64B) padded to 80B -> b128-friendly, bank-spread
#define BLOCK     256
#define GRID_PER_IMG 256

union HPack2 { _Float16 h[2]; unsigned u; };
union BVec   { uint4 q[2]; v16h v; };

// Per block-iteration: 32 pixels x 256 bins of normalized Gaussian weights,
// reduced over pixels with v_wmma_f32_16x16x32_f16 (A = ones).
__global__ __launch_bounds__(BLOCK) void hist_kernel(
    const float* __restrict__ img0, const float* __restrict__ img1,
    const float* __restrict__ img2, float* __restrict__ hist_base,
    int N, int iters)
{
  __shared__ __align__(16) _Float16 tile[BINS * ROW_H]; // [bin][pixel], padded rows
  __shared__ float    sarr[8][TILE_PIX];                // per-chunk partial bin-sums per pixel
  __shared__ __align__(16) unsigned wpack[16];          // packed f16 pair weights (pixel 2q, 2q+1)

  const int t    = threadIdx.x;
  const int lane = t & 31;
  const int wv   = t >> 5;
  const int p    = t & 31;   // pixel slot this thread marches
  const int r    = t >> 5;   // 32-bin chunk this thread marches

  const int img = blockIdx.y;
  const float* __restrict__ x_ptr = (img == 0) ? img0 : (img == 1) ? img1 : img2;
  float* __restrict__ hist = hist_base + img * BINS;

  const float D     = 1.0f / 255.0f;   // bin spacing
  const float INVS2 = 10000.0f;        // 1/sigma^2
  const float R     = __expf(-INVS2 * D * D); // per-bin ratio decay (constant)

  v16h onesA;
  #pragma unroll
  for (int i = 0; i < 16; ++i) onesA[i] = (_Float16)1.0f;

  v8f accA = {0.f,0.f,0.f,0.f,0.f,0.f,0.f,0.f};
  v8f accB = {0.f,0.f,0.f,0.f,0.f,0.f,0.f,0.f};

  const int g0   = 2 * wv;       // this wave's bin-groups (16 bins each)
  const int g1   = 2 * wv + 1;
  const int nb   = lane & 15;    // B-matrix column within group
  const int half = lane >> 4;    // K-range selector per ISA B layout

  for (int it = 0; it < iters; ++it) {
    // ---------------- march phase: fill 32x256 weight tile ----------------
    int pix = blockIdx.x * iters + it;
    int idx = pix * TILE_PIX + p;
    bool valid = (idx < N);
    float x = valid ? x_ptr[idx] : 0.0f;

    const int jlo = r * 32, jhi = jlo + 31;
    int j0 = (int)rintf(x * 255.0f);
    int jc = min(max(j0, jlo), jhi);       // chunk end nearest the Gaussian peak

    float xc = x - (float)jc * D;
    float e0 = valid ? __expf(-0.5f * INVS2 * xc * xc) : 0.0f;
    float f  = __expf( INVS2 * D * (xc - 0.5f * D));   // up-march ratio
    float g  = __expf(-INVS2 * D * (xc + 0.5f * D));   // down-march ratio

    float s_part = 0.0f;
    float e = e0;
    for (int j = jc; j <= jhi; ++j) {      // march up, away from / across peak
      s_part += e;
      tile[j * ROW_H + p] = (_Float16)e;
      e *= f; f *= R;
    }
    e = e0 * g; g *= R;
    for (int j = jc - 1; j >= jlo; --j) {  // march down
      s_part += e;
      tile[j * ROW_H + p] = (_Float16)e;
      e *= g; g *= R;
    }
    sarr[r][p] = s_part;
    __syncthreads();

    // ------------- per-pixel normalization weights, packed f16 -------------
    if (t < 16) {
      float s0 = 1e-8f, s1 = 1e-8f;
      #pragma unroll
      for (int rr = 0; rr < 8; ++rr) { s0 += sarr[rr][2*t]; s1 += sarr[rr][2*t+1]; }
      HPack2 u;
      u.h[0] = (_Float16)fminf(1.0f / s0, 60000.0f);
      u.h[1] = (_Float16)fminf(1.0f / s1, 60000.0f);
      wpack[t] = u.u;
    }
    __syncthreads();

    // ---------------- WMMA reduce phase: D += ones x (B .* w) ----------------
    BVec wvec;
    {
      const uint4* w4 = (const uint4*)wpack;
      wvec.q[0] = w4[half];        // pixels 0..7   or 8..15
      wvec.q[1] = w4[2 + half];    // pixels 16..23 or 24..31
    }
    {
      BVec bm;
      const uint4* rowp = (const uint4*)&tile[(g0 * 16 + nb) * ROW_H];
      bm.q[0] = rowp[half];        // ds_load_b128: K pixel groups per ISA layout
      bm.q[1] = rowp[2 + half];
      v16h bs = bm.v * wvec.v;     // v_pk_mul_f16: fold per-pixel normalization
      accA = __builtin_amdgcn_wmma_f32_16x16x32_f16(false, onesA, false, bs,
                                                    (short)0, accA, false, false);
    }
    {
      BVec bm;
      const uint4* rowp = (const uint4*)&tile[(g1 * 16 + nb) * ROW_H];
      bm.q[0] = rowp[half];
      bm.q[1] = rowp[2 + half];
      v16h bs = bm.v * wvec.v;
      accB = __builtin_amdgcn_wmma_f32_16x16x32_f16(false, onesA, false, bs,
                                                    (short)0, accB, false, false);
    }
    __syncthreads();  // tile reused next iteration
  }

  // D row 0 (VGPR0, lanes 0..15) holds the per-bin column sums
  if (lane < 16) {
    atomicAdd(&hist[g0 * 16 + lane], accA[0]);
    atomicAdd(&hist[g1 * 16 + lane], accB[0]);
  }
}

__global__ void init_ws(float* __restrict__ ws) {
  int t = threadIdx.x;
  if (t < 3 * BINS) ws[t] = 0.0f;
}

__global__ __launch_bounds__(256) void loss_kernel(const float* __restrict__ ws,
                                                   float* __restrict__ out) {
  __shared__ float red[256];
  int t = threadIdx.x;
  float hf = ws[t], hi = ws[BINS + t], hv = ws[2 * BINS + t];
  float a = hf - hi, b = hf - hv;
  red[t] = (0.5f / (float)BINS) * (a * a + b * b);  // alpha=0.5 on both MSE means
  __syncthreads();
  for (int s = 128; s > 0; s >>= 1) {
    if (t < s) red[t] += red[t + s];
    __syncthreads();
  }
  if (t == 0) out[0] = red[0];
}

extern "C" void kernel_launch(void* const* d_in, const int* in_sizes, int n_in,
                              void* d_out, int out_size, void* d_ws, size_t ws_size,
                              hipStream_t stream) {
  const float* fused = (const float*)d_in[0];
  const float* ir    = (const float*)d_in[1];
  const float* vis   = (const float*)d_in[2];
  float* ws  = (float*)d_ws;
  float* out = (float*)d_out;

  int N = in_sizes[0];
  int gridx = GRID_PER_IMG;
  int iters = (N + TILE_PIX * gridx - 1) / (TILE_PIX * gridx);

  hipLaunchKernelGGL(init_ws, dim3(1), dim3(768), 0, stream, ws);
  hipLaunchKernelGGL(hist_kernel, dim3(gridx, 3), dim3(BLOCK), 0, stream,
                     fused, ir, vis, ws, N, iters);
  hipLaunchKernelGGL(loss_kernel, dim3(1), dim3(256), 0, stream, ws, out);
}
